// CLIPVisionTower_llavaprumerge_31078383354519
// MI455X (gfx1250) — compile-verified, matrix operand order checked by reading, not run on hardware
//
#include <hip/hip_runtime.h>
#include <hip/hip_bf16.h>
#include <stdint.h>

// ---------------------------------------------------------------------------
// Problem constants (reference: B=32, N=576, C=1024, L=N/8=72, K_CLUSTER=32)
// ---------------------------------------------------------------------------
#define BB   32
#define NTOK 576          // tokens (without CLS)
#define NP1  577          // N + 1 (with CLS)
#define CC   1024
#define LL   72           // top-k kept tokens
#define KCL  32           // cluster size
#define NCOMPL (NTOK - LL)   // 504
#define MT_PAD 80         // 72 padded to 5 x 16 WMMA tiles
#define NEG_INF (-3.402823466e38f)

typedef __attribute__((ext_vector_type(16))) _Float16 v16h;
typedef __attribute__((ext_vector_type(8)))  _Float16 v8h;
typedef __attribute__((ext_vector_type(8)))  float    v8f;

// ---------------------------------------------------------------------------
// Kernel 1: logits = (q_cls . keys) * C^-0.5 ; softmax over 577 ; drop CLS.
// One block per batch. q_cls staged in LDS; one wave per key row.
// ---------------------------------------------------------------------------
__global__ void k1_attn_softmax(const float* __restrict__ queries,
                                const float* __restrict__ keys,
                                float* __restrict__ cls_attn) {
    __shared__ float s_q[CC];
    __shared__ float s_logit[NP1];
    __shared__ float s_red[256];

    const int b    = blockIdx.x;
    const int tid  = threadIdx.x;
    const int lane = tid & 31;
    const int wave = tid >> 5;
    const float scale = 0.03125f; // 1024^-0.5

    const float* q = queries + (size_t)b * NP1 * CC; // row 0 = CLS query
    for (int c = tid; c < CC; c += 256) s_q[c] = q[c];
    __syncthreads();

    // 8 waves, each handles rows n = wave, wave+8, ...
    for (int n = wave; n < NP1; n += 8) {
        const float* krow = keys + ((size_t)b * NP1 + n) * CC;
        float p = 0.f;
        for (int c = lane; c < CC; c += 32) p += krow[c] * s_q[c];
        #pragma unroll
        for (int off = 16; off >= 1; off >>= 1) p += __shfl_down(p, off, 32);
        if (lane == 0) s_logit[n] = p * scale;
    }
    __syncthreads();

    // block max
    float m = NEG_INF;
    for (int n = tid; n < NP1; n += 256) m = fmaxf(m, s_logit[n]);
    s_red[tid] = m; __syncthreads();
    for (int s = 128; s > 0; s >>= 1) {
        if (tid < s) s_red[tid] = fmaxf(s_red[tid], s_red[tid + s]);
        __syncthreads();
    }
    const float mx = s_red[0];
    __syncthreads();

    // exp + block sum
    float sum = 0.f;
    for (int n = tid; n < NP1; n += 256) {
        float e = expf(s_logit[n] - mx);
        s_logit[n] = e;
        sum += e;
    }
    s_red[tid] = sum; __syncthreads();
    for (int s = 128; s > 0; s >>= 1) {
        if (tid < s) s_red[tid] += s_red[tid + s];
        __syncthreads();
    }
    const float inv = 1.f / s_red[0];
    __syncthreads();

    for (int n = 1 + tid; n < NP1; n += 256)
        cls_attn[(size_t)b * NTOK + (n - 1)] = s_logit[n] * inv;
}

// ---------------------------------------------------------------------------
// Kernel 2: exact top-L by rank counting (ties -> lower index wins, matching
// jax.lax.top_k), complement in ascending index order.
// perm[b, 0..71]  = topk indices, descending value order
// perm[b, 72..575]= complement indices, ascending
// all_attn[b, j]  = cls_attn[b, perm[b, j]]
// One block (576 threads) per batch.
// ---------------------------------------------------------------------------
__global__ void k2_topk_perm(const float* __restrict__ cls_attn,
                             int* __restrict__ perm,
                             float* __restrict__ all_attn) {
    __shared__ float s_a[NTOK];
    __shared__ int   s_top[NTOK];

    const int b = blockIdx.x;
    const int n = threadIdx.x;   // blockDim.x == 576

    s_a[n] = cls_attn[(size_t)b * NTOK + n];
    __syncthreads();

    const float an = s_a[n];
    int rank = 0;
    for (int m2 = 0; m2 < NTOK; ++m2) {
        float am = s_a[m2];
        rank += (am > an) || (am == an && m2 < n);
    }
    s_top[n] = (rank < LL) ? 1 : 0;
    __syncthreads();

    if (rank < LL) {
        perm[(size_t)b * NTOK + rank]     = n;
        all_attn[(size_t)b * NTOK + rank] = an;
    } else {
        int pos = 0;
        for (int m2 = 0; m2 < n; ++m2) pos += (s_top[m2] == 0);
        perm[(size_t)b * NTOK + LL + pos]     = n;
        all_attn[(size_t)b * NTOK + LL + pos] = an;
    }
}

// ---------------------------------------------------------------------------
// Kernel 3: AK[b, j, :] = f16( l2norm( keys[b, 1 + perm[b,j], :] ) )
// One block (256 threads) per (b, j) row.
// ---------------------------------------------------------------------------
__global__ void k3_normalize_gather(const float* __restrict__ keys,
                                    const int* __restrict__ perm,
                                    _Float16* __restrict__ AK) {
    __shared__ float s_red[256];
    const int row = blockIdx.x;          // b*576 + j
    const int b   = row / NTOK;
    const int tid = threadIdx.x;

    const int tok = perm[row];
    const float* src = keys + ((size_t)b * NP1 + 1 + tok) * CC;

    float ss = 0.f;
    for (int c = tid; c < CC; c += 256) { float v = src[c]; ss += v * v; }
    s_red[tid] = ss; __syncthreads();
    for (int s = 128; s > 0; s >>= 1) {
        if (tid < s) s_red[tid] += s_red[tid + s];
        __syncthreads();
    }
    const float inv = 1.f / sqrtf(s_red[0] + 1e-12f);

    _Float16* dst = AK + (size_t)row * CC;
    for (int c = tid; c < CC; c += 256) dst[c] = (_Float16)(src[c] * inv);
}

// ---------------------------------------------------------------------------
// Kernel 4: sim[b] = AK[b,0:72] @ AK[b]^T  via v_wmma_f32_16x16x32_f16.
// One wave per 16x16 tile, K-loop 1024/32. M padded 72->80 (rows clamped).
// Diagonal (l==j, l<72) masked to -FLT_MAX in the epilogue.
// A layout: M = lane&15, K-half = lane>>4; two b128 loads per K-step.
// B layout mirrors A with N = lane&15 (B rows are also key vectors).
// ---------------------------------------------------------------------------
__global__ void k4_sim_wmma(const _Float16* __restrict__ AK,
                            float* __restrict__ sim) {
    const int tid  = threadIdx.x;
    const int lane = tid & 31;
    const int wave = tid >> 5;

    const int t  = blockIdx.x * 8 + wave;        // tile id, grid sized exactly
    const int b  = t / (5 * 36);
    const int r  = t % (5 * 36);
    const int mt = r / 36;
    const int nt = r % 36;

    const int mrow0 = mt * 16;
    const int ncol0 = nt * 16;

    int arow = mrow0 + (lane & 15);
    if (arow > LL - 1) arow = LL - 1;            // clamp padded rows
    const int ncol = ncol0 + (lane & 15);
    const int h8   = (lane >> 4) * 8;            // K-half offset

    const _Float16* Ap = AK + ((size_t)b * NTOK + arow) * CC;
    const _Float16* Bp = AK + ((size_t)b * NTOK + ncol) * CC;

    v8f acc = {};
    for (int kk = 0; kk < CC; kk += 32) {
        v8h alo = *(const v8h*)(Ap + kk + h8);
        v8h ahi = *(const v8h*)(Ap + kk + 16 + h8);
        v8h blo = *(const v8h*)(Bp + kk + h8);
        v8h bhi = *(const v8h*)(Bp + kk + 16 + h8);
        v16h a = __builtin_shufflevector(alo, ahi,
                  0,1,2,3,4,5,6,7,8,9,10,11,12,13,14,15);
        v16h bm = __builtin_shufflevector(blo, bhi,
                  0,1,2,3,4,5,6,7,8,9,10,11,12,13,14,15);
        if (kk + 32 < CC) {                      // uniform: no EXEC divergence
            __builtin_prefetch(Ap + kk + 32, 0, 3);
            __builtin_prefetch(Bp + kk + 32, 0, 3);
        }
        acc = __builtin_amdgcn_wmma_f32_16x16x32_f16(
                  false, a, false, bm, (short)0, acc, false, false);
    }

    // Epilogue: C/D layout: VGPR v -> M = v + (lane>=16 ? 8 : 0), N = lane&15
    const int mofs = (lane >> 4) * 8;
    #pragma unroll
    for (int v = 0; v < 8; ++v) {
        const int M = mrow0 + v + mofs;
        if (M < LL) {
            const int Ncol = ncol0 + (lane & 15);
            float val = acc[v];
            if (M == Ncol) val = NEG_INF;        // diagonal mask
            sim[((size_t)b * LL + M) * NTOK + Ncol] = val;
        }
    }
}

// ---------------------------------------------------------------------------
// Kernel 5: per (b, l): top-32 of sim row (iterative argmax, tie->lowest idx),
// merged[b,l,:] = sum_k w_k * image[b, perm[b,cidx_k], :] + image[b, perm[b,l], :]
// One block (256 threads) per row.
// ---------------------------------------------------------------------------
__global__ void k5_cluster_merge(const float* __restrict__ sim,
                                 const int* __restrict__ perm,
                                 const float* __restrict__ all_attn,
                                 const float* __restrict__ image,
                                 float* __restrict__ out) {
    __shared__ float s_val[NTOK];
    __shared__ float s_rv[256];
    __shared__ int   s_ri[256];
    __shared__ int   s_rows[KCL];
    __shared__ float s_w[KCL];

    const int rowid = blockIdx.x;   // b*72 + l
    const int b   = rowid / LL;
    const int l   = rowid % LL;
    const int tid = threadIdx.x;

    const float* srow = sim + (size_t)rowid * NTOK;
    for (int n = tid; n < NTOK; n += 256) s_val[n] = srow[n];
    __syncthreads();

    for (int it = 0; it < KCL; ++it) {
        float best = NEG_INF; int bi = NTOK;
        for (int n = tid; n < NTOK; n += 256) {
            float v = s_val[n];
            if (v > best || (v == best && n < bi)) { best = v; bi = n; }
        }
        s_rv[tid] = best; s_ri[tid] = bi;
        __syncthreads();
        for (int s = 128; s > 0; s >>= 1) {
            if (tid < s) {
                float ov = s_rv[tid + s]; int oi = s_ri[tid + s];
                if (ov > s_rv[tid] || (ov == s_rv[tid] && oi < s_ri[tid])) {
                    s_rv[tid] = ov; s_ri[tid] = oi;
                }
            }
            __syncthreads();
        }
        if (tid == 0) {
            const int sel = s_ri[0];
            s_rows[it] = perm[(size_t)b * NTOK + sel];
            s_w[it]    = all_attn[(size_t)b * NTOK + sel];
            s_val[sel] = NEG_INF;
        }
        __syncthreads();
    }

    const int self = perm[(size_t)b * NTOK + l];
    const float* imgb = image + (size_t)b * NTOK * CC;
    float* orow = out + ((size_t)b * (LL + 1) + l) * CC;

    for (int c = tid; c < CC; c += 256) {
        float acc = imgb[(size_t)self * CC + c];   // residual x_others
        #pragma unroll 4
        for (int k2 = 0; k2 < KCL; ++k2)
            acc += s_w[k2] * imgb[(size_t)s_rows[k2] * CC + c];
        orow[c] = acc;
    }
}

// ---------------------------------------------------------------------------
// Kernel 6: extra row = sum over complement of attn-weighted image rows.
// One block (256 threads) per batch; 4 columns per thread.
// ---------------------------------------------------------------------------
__global__ void k6_extra(const int* __restrict__ perm,
                         const float* __restrict__ all_attn,
                         const float* __restrict__ image,
                         float* __restrict__ out) {
    const int b   = blockIdx.x;
    const int tid = threadIdx.x;

    const float* imgb = image + (size_t)b * NTOK * CC;
    float acc[4] = {0.f, 0.f, 0.f, 0.f};

    for (int j = 0; j < NCOMPL; ++j) {
        const int tok = perm[(size_t)b * NTOK + LL + j];
        const float w = all_attn[(size_t)b * NTOK + LL + j];
        const float* row = imgb + (size_t)tok * CC;
        #pragma unroll
        for (int q = 0; q < 4; ++q) acc[q] += w * row[tid + q * 256];
    }
    float* orow = out + ((size_t)b * (LL + 1) + LL) * CC;
    #pragma unroll
    for (int q = 0; q < 4; ++q) orow[tid + q * 256] = acc[q];
}

// ---------------------------------------------------------------------------
// Host-side launcher. Inputs: [0]=image_features (B,576,1024) f32,
// [1]=keys (B,577,1024) f32, [2]=queries (B,577,1024) f32.
// Output: (B,73,1024) f32.
// ---------------------------------------------------------------------------
extern "C" void kernel_launch(void* const* d_in, const int* in_sizes, int n_in,
                              void* d_out, int out_size, void* d_ws, size_t ws_size,
                              hipStream_t stream) {
    const float* image   = (const float*)d_in[0];
    const float* keys    = (const float*)d_in[1];
    const float* queries = (const float*)d_in[2];
    float* out = (float*)d_out;

    // Workspace layout (all chunks multiples of 256 bytes)
    char* wsp = (char*)d_ws;
    size_t off = 0;
    float*    cls_attn = (float*)(wsp + off);    off += (size_t)BB * NTOK * sizeof(float);
    int*      perm     = (int*)(wsp + off);      off += (size_t)BB * NTOK * sizeof(int);
    float*    all_attn = (float*)(wsp + off);    off += (size_t)BB * NTOK * sizeof(float);
    _Float16* AK       = (_Float16*)(wsp + off); off += (size_t)BB * NTOK * CC * sizeof(_Float16);
    float*    sim      = (float*)(wsp + off);    off += (size_t)BB * LL * NTOK * sizeof(float);
    (void)ws_size; (void)in_sizes; (void)n_in; (void)out_size;

    // 1) CLS attention softmax
    k1_attn_softmax<<<BB, 256, 0, stream>>>(queries, keys, cls_attn);

    // 2) exact top-L permutation + gathered attention
    k2_topk_perm<<<BB, NTOK, 0, stream>>>(cls_attn, perm, all_attn);

    // 3) l2-normalized, permuted keys in f16
    k3_normalize_gather<<<BB * NTOK, 256, 0, stream>>>(keys, perm, AK);

    // 4) sim GEMM via WMMA: tiles = B * (80/16) * (576/16) = 32*5*36 = 5760
    //    8 waves per block -> 720 blocks exactly
    k4_sim_wmma<<<(BB * (MT_PAD / 16) * (NTOK / 16)) / 8, 256, 0, stream>>>(AK, sim);

    // 5) per-row top-32 + weighted merge (+ residual)
    k5_cluster_merge<<<BB * LL, 256, 0, stream>>>(sim, perm, all_attn, image, out);

    // 6) extra row
    k6_extra<<<BB, 256, 0, stream>>>(perm, all_attn, image, out);
}
